// GVPTransCond_43774306681166
// MI455X (gfx1250) — matrix operand order repaired and checked by reading, not compiled, not run on hardware
//
#include <hip/hip_runtime.h>
#include <hip/hip_bf16.h>
#include <math.h>

// ---------------------------------------------------------------------------
// Problem constants (from the reference)
// ---------------------------------------------------------------------------
#define B_    16
#define L_    512
#define Nn    (B_*L_)        // 8192 nodes
#define Ee    (Nn*30)        // 245760 edges
#define NS    128
#define NV    16
#define ESD   32
#define HEADS 8
#define HD    (NS/HEADS)     // 16
#define FFN_  512
#define OUTD  4
#define TOK   (L_*B_)        // 8192 tokens
#define CE    30720          // edge chunk size
#define NCH   (Ee/CE)        // 8 chunks
#define EPSV  1e-8f
#define LNE   1e-5f

typedef __attribute__((ext_vector_type(16))) _Float16 v16h;
typedef __attribute__((ext_vector_type(8)))  _Float16 v8h;
typedef __attribute__((ext_vector_type(8)))  float    v8f;

// ---------------------------------------------------------------------------
// Generic batched strided WMMA GEMM:  C = act( oscale*(A @ B' + bias) )
//  A:  f32, element A[z*sA + m*lda + k]
//  B:  BMODE==0 : B[z*sB + k*ldb + n]   (K x N)
//      BMODE==1 : B[z*sB + n*ldb + k]   (N x K, i.e. jax weight [dout,din])
//  C:  C[z*sC + m*ldc + n]
//
// Block tile 64x32, 8 waves (4x2 of 16x16), K stepped by 32.
// Fragments are staged in LDS pre-permuted into the CDNA5 WMMA VGPR layout:
//   A frag: lane = (m&15)|(k[3]<<4), half = k[2:0]|(k[4]<<3)
//   B frag: lane = (n&15)|(k[3]<<4), half = k[2:0]|(k[4]<<3)
// Key property: for a fixed LDS lane, each aligned 8-half run maps to 8
// CONTIGUOUS k values, so every staging thread converts 8 contiguous floats
// and issues a single aligned ds_store_b128. Bounds handling is branch-free:
// unconditional loads from clamped addresses + cndmask zero-select.
// ---------------------------------------------------------------------------
template<int BMODE, int ACT>
__global__ __launch_bounds__(256)
void k_gemm(const float* __restrict__ A, long lda, long sA,
            const float* __restrict__ Bm, long ldb, long sB,
            const float* __restrict__ bias,
            float* __restrict__ C, long ldc, long sC,
            int M, int K, int Nout, float oscale)
{
  __shared__ alignas(32) _Float16 lsA[64*32];
  __shared__ alignas(32) _Float16 lsB[32*32];
  const int t    = threadIdx.x;
  const int lane = t & 31;
  const int wave = t >> 5;
  const int wm   = wave >> 1;           // 0..3 (M group)
  const int wn   = wave & 1;            // 0..1 (N group)
  const int m0   = blockIdx.x * 64;
  const int n0   = blockIdx.y * 32;
  const long z   = blockIdx.z;
  const float* Ab = A  + z * sA;
  const float* Bb = Bm + z * sB;
  float*       Cb = C  + z * sC;

  // ---- per-thread staging geometry (constant across K loop) ---------------
  // A: thread t -> group g (16 rows), lds-lane la, half-block hba (0/1)
  const int ga   = t >> 6;              // 0..3
  const int ra   = t & 63;
  const int la   = ra >> 1;             // 0..31
  const int hba  = ra & 1;              // 0..1
  const int gmA  = m0 + ga * 16 + (la & 15);
  const int kA0  = hba * 16 + ((la >> 4) << 3);   // + kb + j
  const int mA   = (gmA < M) ? gmA : (M - 1);
  const bool mAv = (gmA < M);
  _Float16* stA  = &lsA[((ga * 32 + la) << 4) + hba * 8];
  // B: threads 0..127 -> group ngb (16 cols), lds-lane lb, half-block hbb
  const int ngb  = (t >> 6) & 1;        // 0..1
  const int lb   = la;                  // reuse decomposition
  const int hbb  = hba;
  const int gnB  = n0 + ngb * 16 + (lb & 15);
  const int kB0  = hbb * 16 + ((lb >> 4) << 3);   // + kb + j
  const int nB   = (gnB < Nout) ? gnB : (Nout - 1);
  const bool nBv = (gnB < Nout);
  _Float16* stB  = &lsB[((ngb * 32 + lb) << 4) + hbb * 8];

  v8f acc = {};
  const int ksteps = (K + 31) >> 5;
  for (int ks = 0; ks < ksteps; ++ks) {
    const int kb = ks << 5;
    __syncthreads();
    // ---- stage A: 8 contiguous k floats -> one ds_store_b128 --------------
    {
      const float* ap = Ab + (long)mA * lda;
      v8h h;
#pragma unroll
      for (int j = 0; j < 8; ++j) {
        int k = kb + kA0 + j;
        int kc = (k < K) ? k : (K - 1);
        float v = ap[kc];
        v = (mAv && k < K) ? v : 0.f;
        h[j] = (_Float16)v;
      }
      *(v8h*)stA = h;
    }
    // ---- stage B (threads 0..127): one ds_store_b128 ----------------------
    if (t < 128) {
      v8h h;
#pragma unroll
      for (int j = 0; j < 8; ++j) {
        int k = kb + kB0 + j;
        int kc = (k < K) ? k : (K - 1);
        float v = (BMODE == 0) ? Bb[(long)kc * ldb + nB]
                               : Bb[(long)nB * ldb + kc];
        v = (nBv && k < K) ? v : 0.f;
        h[j] = (_Float16)v;
      }
      *(v8h*)stB = h;
    }
    // speculative prefetch of next A k-slab (global_prefetch_b8)
    if (ks + 1 < ksteps)
      __builtin_prefetch(&Ab[(long)mA * lda + kb + 32 + kA0], 0, 1);
    __syncthreads();
    v16h af = *(const v16h*)&lsA[(wm * 32 + lane) << 4];
    v16h bf = *(const v16h*)&lsB[(wn * 32 + lane) << 4];
    acc = __builtin_amdgcn_wmma_f32_16x16x32_f16(false, af, false, bf,
                                                 (short)0, acc, false, false);
  }
  // ---- epilogue: D layout VGPR r -> M = r + 8*(lane>=16), N = lane&15 -----
#pragma unroll
  for (int r = 0; r < 8; ++r) {
    int gm = m0 + wm * 16 + r + ((lane >> 4) << 3);
    int gn = n0 + wn * 16 + (lane & 15);
    if (gm < M && gn < Nout) {
      float v = acc[r];
      if (bias) v += bias[gn];
      v *= oscale;
      if (ACT == 1) v = v > 0.f ? v : 0.f;
      Cb[(long)gm * ldc + gn] = v;
    }
  }
}

// ---------------------------------------------------------------------------
// Elementwise / gather / reduction kernels
// ---------------------------------------------------------------------------
__global__ void k_zero(float* p, long n) {
  long t = (long)blockIdx.x * 256 + threadIdx.x;
  if (t < n) p[t] = 0.f;
}

__global__ void k_add(float* x, const float* y, long n) {
  long t = (long)blockIdx.x * 256 + threadIdx.x;
  if (t < n) x[t] += y[t];
}

// init: ln8 of [z_t,0] into X24 cols 0..7 ; node_v normalized & transposed
__global__ void k_node_init(const float* __restrict__ z, const float* __restrict__ nv,
                            const float* __restrict__ lnw, const float* __restrict__ lnb,
                            float* __restrict__ X24, float* __restrict__ vT4)
{
  int n = blockIdx.x * 256 + threadIdx.x;
  if (n >= Nn) return;
  float s[8];
  float mean = 0.f;
  for (int c = 0; c < 8; ++c) { s[c] = (c < 4) ? z[n*4 + c] : 0.f; mean += s[c]; }
  mean *= 0.125f;
  float var = 0.f;
  for (int c = 0; c < 8; ++c) { float d = s[c] - mean; var += d * d; }
  var *= 0.125f;
  float rs = rsqrtf(var + LNE);
  for (int c = 0; c < 8; ++c) X24[n*24 + c] = (s[c] - mean) * rs * lnw[c] + lnb[c];
  // gvp_ln on vectors: vn = sqrt(mean_ch(clip(sumsq,EPS)))
  float acc = 0.f;
  for (int j = 0; j < 4; ++j) {
    float ss = 0.f;
    for (int d = 0; d < 3; ++d) { float v = nv[n*12 + j*3 + d]; ss += v * v; }
    acc += fmaxf(ss, EPSV);
  }
  float sc = rsqrtf(acc * 0.25f);
  for (int d = 0; d < 3; ++d)
    for (int j = 0; j < 4; ++j)
      vT4[n*12 + d*4 + j] = nv[n*12 + j*3 + d] * sc;
}

// vnorm over the "3" axis of vh [M,3,H] -> out[m, col0+j] (row stride ldo)
__global__ void k_vnorm3(const float* __restrict__ vh, float* __restrict__ out,
                         long ldo, int col0, int M, int H)
{
  long t = (long)blockIdx.x * 256 + threadIdx.x;
  if (t >= (long)M * H) return;
  long m = t / H; int j = (int)(t % H);
  float ss = 0.f;
  for (int d = 0; d < 3; ++d) { float v = vh[m*3*H + (long)d*H + j]; ss += v * v; }
  out[m * ldo + col0 + j] = sqrtf(fmaxf(ss, EPSV));
}

// vt[M,3,H] *= sigmoid(gate[M,H]) broadcast over the 3 axis
__global__ void k_gate_mul(float* __restrict__ vt, const float* __restrict__ gl,
                           int M, int H)
{
  long t = (long)blockIdx.x * 256 + threadIdx.x;
  if (t >= (long)M * 3 * H) return;
  long m = t / (3 * H); int j = (int)(t % H);
  float g = 1.f / (1.f + expf(-gl[m*H + j]));
  vt[t] *= g;
}

// [M,3,H] -> [M,H,3]
__global__ void k_transpose_out(const float* __restrict__ vt, float* __restrict__ dst,
                                int M, int H)
{
  long t = (long)blockIdx.x * 256 + threadIdx.x;
  if (t >= (long)M * 3 * H) return;
  long m = t / (3 * H); int r = (int)(t % (3 * H));
  int d = r / H, j = r % H;
  dst[m*3*H + (long)j*3 + d] = vt[t];
}

// [M,H,3] -> [M,3,H]
__global__ void k_transpose_in(const float* __restrict__ v, float* __restrict__ vT,
                               int M, int H)
{
  long t = (long)blockIdx.x * 256 + threadIdx.x;
  if (t >= (long)M * 3 * H) return;
  long m = t / (3 * H); int r = (int)(t % (3 * H));
  int d = r / H, j = r % H;
  vT[t] = v[m*3*H + (long)j*3 + d];
}

__global__ void k_copy_cols(const float* __restrict__ S, int scols,
                            float* __restrict__ X, long ld, int col0, int M)
{
  long t = (long)blockIdx.x * 256 + threadIdx.x;
  if (t >= (long)M * scols) return;
  long m = t / scols; int c = (int)(t % scols);
  X[m * ld + col0 + c] = S[t];
}

__global__ void k_relu_cols(float* __restrict__ X, long ld, int ncols, int M)
{
  long t = (long)blockIdx.x * 256 + threadIdx.x;
  if (t >= (long)M * ncols) return;
  long m = t / ncols; int c = (int)(t % ncols);
  float v = X[m * ld + c];
  X[m * ld + c] = v > 0.f ? v : 0.f;
}

// edge init: ln32(edge_s) -> Xe[:,0..31]; Xe[:,32]=||vh||; evh = (v/||v||)*wh00
__global__ void k_edge_init(const float* __restrict__ es, const float* __restrict__ ev,
                            const float* __restrict__ lnw, const float* __restrict__ lnb,
                            const float* __restrict__ wh, float* __restrict__ Xe,
                            float* __restrict__ evh)
{
  long e = (long)blockIdx.x * 256 + threadIdx.x;
  if (e >= Ee) return;
  float mean = 0.f;
  for (int c = 0; c < 32; ++c) mean += es[e*32 + c];
  mean *= (1.f / 32.f);
  float var = 0.f;
  for (int c = 0; c < 32; ++c) { float d = es[e*32 + c] - mean; var += d * d; }
  var *= (1.f / 32.f);
  float rs = rsqrtf(var + LNE);
  for (int c = 0; c < 32; ++c)
    Xe[e*33 + c] = (es[e*32 + c] - mean) * rs * lnw[c] + lnb[c];
  float ss = 0.f;
  for (int d = 0; d < 3; ++d) { float v = ev[e*3 + d]; ss += v * v; }
  float sc = rsqrtf(fmaxf(ss, EPSV));
  float w0 = wh[0];
  float s2 = 0.f;
  for (int d = 0; d < 3; ++d) {
    float v = ev[e*3 + d] * sc * w0;
    evh[e*3 + d] = v;
    s2 += v * v;
  }
  Xe[e*33 + 32] = sqrtf(fmaxf(s2, EPSV));
}

// edge vector gate: evb *= wv00 * sigmoid(wsv.b + es_out . wsv.w)
__global__ void k_edge_gate(const float* __restrict__ esb, const float* __restrict__ wsvw,
                            const float* __restrict__ wsvb, const float* __restrict__ wv,
                            float* __restrict__ evb)
{
  long e = (long)blockIdx.x * 256 + threadIdx.x;
  if (e >= Ee) return;
  float a = wsvb[0];
  for (int c = 0; c < 32; ++c) a += esb[e*32 + c] * wsvw[c];
  float g = wv[0] / (1.f + expf(-a));
  for (int d = 0; d < 3; ++d) evb[e*3 + d] *= g;
}

__global__ void k_count(const int* __restrict__ dst, float* __restrict__ cnt)
{
  long e = (long)blockIdx.x * 256 + threadIdx.x;
  if (e >= Ee) return;
  atomicAdd(&cnt[dst[e]], 1.f);
}

// build X[:,0..287] = [s[src], es, s[dst]] for one chunk (ld = 321)
__global__ void k_gather_s(const int* __restrict__ src, const int* __restrict__ dst,
                           const float* __restrict__ s, const float* __restrict__ esc,
                           float* __restrict__ X, int C)
{
  long t = (long)blockIdx.x * 256 + threadIdx.x;
  if (t >= (long)C * 288) return;
  long i = t / 288; int c = (int)(t % 288);
  float v;
  if (c < 128)      v = s[(long)src[i]*NS + c];
  else if (c < 160) v = esc[i*32 + (c - 128)];
  else              v = s[(long)dst[i]*NS + (c - 160)];
  X[i * 321 + c] = v;
}

// build mvT [C,3,33] = transpose(concat([v[src], ev, v[dst]], ch))
__global__ void k_gather_v(const int* __restrict__ src, const int* __restrict__ dst,
                           const float* __restrict__ v, const float* __restrict__ evc,
                           float* __restrict__ mvT, int C)
{
  long t = (long)blockIdx.x * 256 + threadIdx.x;
  if (t >= (long)C * 99) return;
  long i = t / 99; int r = (int)(t % 99);
  int d = r / 33, j = r % 33;
  float val;
  if (j < 16)       val = v[(long)src[i]*48 + j*3 + d];
  else if (j == 16) val = evc[i*3 + d];
  else              val = v[(long)dst[i]*48 + (j - 17)*3 + d];
  mvT[i*99 + (long)d*33 + j] = val;
}

__global__ void k_scat_s(const float* __restrict__ sm, const int* __restrict__ dst,
                         float* __restrict__ aggS, int C)
{
  long t = (long)blockIdx.x * 256 + threadIdx.x;
  if (t >= (long)C * 128) return;
  long i = t / 128; int c = (int)(t % 128);
  atomicAdd(&aggS[(long)dst[i]*128 + c], sm[t]);
}

__global__ void k_scat_v(const float* __restrict__ mv, const int* __restrict__ dst,
                         float* __restrict__ aggV, int C)
{
  long t = (long)blockIdx.x * 256 + threadIdx.x;
  if (t >= (long)C * 48) return;
  long i = t / 48; int r = (int)(t % 48);
  atomicAdd(&aggV[(long)dst[i]*48 + r], mv[t]);
}

// s,v += dS/c, dV/c ; then gvp_ln (ln over 128 scalars, vector RMS over 16 ch)
__global__ void k_node_upd(float* __restrict__ s, float* __restrict__ v,
                           const float* __restrict__ dS, const float* __restrict__ dV,
                           const float* __restrict__ cnt,
                           const float* __restrict__ lnw, const float* __restrict__ lnb)
{
  int n = blockIdx.x * 256 + threadIdx.x;
  if (n >= Nn) return;
  float inv = 1.f;
  if (cnt) { float c = cnt[n]; if (c < 1.f) c = 1.f; inv = 1.f / c; }
  float mean = 0.f;
  for (int i = 0; i < NS; ++i) {
    float t = s[(long)n*NS + i] + dS[(long)n*NS + i] * inv;
    s[(long)n*NS + i] = t;
    mean += t;
  }
  mean *= (1.f / NS);
  float var = 0.f;
  for (int i = 0; i < NS; ++i) { float d = s[(long)n*NS + i] - mean; var += d * d; }
  var *= (1.f / NS);
  float rs = rsqrtf(var + LNE);
  for (int i = 0; i < NS; ++i)
    s[(long)n*NS + i] = (s[(long)n*NS + i] - mean) * rs * lnw[i] + lnb[i];
  float acc = 0.f;
  for (int j = 0; j < NV; ++j) {
    float ss = 0.f;
    for (int d = 0; d < 3; ++d) {
      float t = v[(long)n*48 + j*3 + d] + dV[(long)n*48 + j*3 + d] * inv;
      v[(long)n*48 + j*3 + d] = t;
      ss += t * t;
    }
    acc += fmaxf(ss, EPSV);
  }
  float sc = rsqrtf(acc * (1.f / NV));
  for (int r = 0; r < 48; ++r) v[(long)n*48 + r] *= sc;
}

__global__ void k_layernorm(float* __restrict__ x, const float* __restrict__ w,
                            const float* __restrict__ b, int rows, int d)
{
  int r = blockIdx.x * 256 + threadIdx.x;
  if (r >= rows) return;
  float mean = 0.f;
  for (int i = 0; i < d; ++i) mean += x[(long)r*d + i];
  mean /= d;
  float var = 0.f;
  for (int i = 0; i < d; ++i) { float t = x[(long)r*d + i] - mean; var += t * t; }
  var /= d;
  float rs = rsqrtf(var + LNE);
  for (int i = 0; i < d; ++i)
    x[(long)r*d + i] = (x[(long)r*d + i] - mean) * rs * w[i] + b[i];
}

__global__ void k_softmax512(float* __restrict__ a, long rows)
{
  long r = (long)blockIdx.x * 256 + threadIdx.x;
  if (r >= rows) return;
  float* p = a + r * 512;
  float mx = p[0];
  for (int i = 1; i < 512; ++i) mx = fmaxf(mx, p[i]);
  float sum = 0.f;
  for (int i = 0; i < 512; ++i) { float e = expf(p[i] - mx); p[i] = e; sum += e; }
  float inv = 1.f / sum;
  for (int i = 0; i < 512; ++i) p[i] *= inv;
}

// x[t=l*B+b, c] = mix_out[b*L+l, c] + positional_encoding(l, c)
__global__ void k_build_x(const float* __restrict__ xb, float* __restrict__ x)
{
  long t = (long)blockIdx.x * 256 + threadIdx.x;
  if (t >= (long)TOK * NS) return;
  long tok = t / NS; int c = (int)(t % NS);
  int l = (int)(tok / B_), b = (int)(tok % B_);
  int j = (c < 64) ? c : (c - 64);
  float fr = expf(-(float)j * (logf(10000.f) / 63.f));
  float ang = (float)l * fr;
  float pe = (c < 64) ? sinf(ang) : cosf(ang);
  x[t] = xb[((long)b*L_ + l)*NS + c] + pe;
}

__global__ void k_add_cond(float* __restrict__ x, const float* __restrict__ tc)
{
  long t = (long)blockIdx.x * 256 + threadIdx.x;
  if (t >= (long)TOK * NS) return;
  long tok = t / NS; int c = (int)(t % NS);
  int b = (int)(tok % B_);
  x[t] += tc[b*NS + c];
}

// time embedding: tc[b] = fc2(silu(fc1([t, sin(2pi t w), cos(2pi t w)])))
__global__ void k_time(const float* __restrict__ noise, const float* __restrict__ w8,
                       const float* __restrict__ f1b, const float* __restrict__ f1w,
                       const float* __restrict__ f2b, const float* __restrict__ f2w,
                       float* __restrict__ tc)
{
  __shared__ float h1[256];
  int b = blockIdx.x;
  int o = threadIdx.x;
  float t = noise[b];
  float acc = f1b[o] + f1w[o*17] * t;
  for (int j = 0; j < 8; ++j) {
    float f = 6.283185307179586f * t * w8[j];
    acc += f1w[o*17 + 1 + j] * sinf(f) + f1w[o*17 + 9 + j] * cosf(f);
  }
  h1[o] = acc / (1.f + expf(-acc));   // silu
  __syncthreads();
  if (o < NS) {
    float a2 = f2b[o];
    for (int j = 0; j < 256; ++j) a2 += f2w[o*256 + j] * h1[j];
    tc[b*NS + o] = a2;
  }
}

// outT [l*B+b, 4] -> out [b*L+l, 4]
__global__ void k_final(const float* __restrict__ oT, float* __restrict__ out)
{
  long t = (long)blockIdx.x * 256 + threadIdx.x;
  if (t >= (long)TOK * OUTD) return;
  long tok = t / OUTD; int c = (int)(t % OUTD);
  int l = (int)(tok / B_), b = (int)(tok % B_);
  out[((long)b*L_ + l)*OUTD + c] = oT[t];
}

// ---------------------------------------------------------------------------
// Host-side orchestration
// ---------------------------------------------------------------------------
static inline unsigned gblk(long n) { return (unsigned)((n + 255) / 256); }

static void gemm(hipStream_t st, int bmode, int act,
                 const float* A, long lda, long sA,
                 const float* Bm, long ldb, long sB,
                 const float* bias, float* C, long ldc, long sC,
                 int M, int K, int Nout, int nb, float oscale)
{
  dim3 g((M + 63) / 64, (Nout + 31) / 32, nb), blk(256);
  if (bmode == 0) {
    if (act == 0) k_gemm<0,0><<<g,blk,0,st>>>(A,lda,sA,Bm,ldb,sB,bias,C,ldc,sC,M,K,Nout,oscale);
    else          k_gemm<0,1><<<g,blk,0,st>>>(A,lda,sA,Bm,ldb,sB,bias,C,ldc,sC,M,K,Nout,oscale);
  } else {
    if (act == 0) k_gemm<1,0><<<g,blk,0,st>>>(A,lda,sA,Bm,ldb,sB,bias,C,ldc,sC,M,K,Nout,oscale);
    else          k_gemm<1,1><<<g,blk,0,st>>>(A,lda,sA,Bm,ldb,sB,bias,C,ldc,sC,M,K,Nout,oscale);
  }
}

extern "C" void kernel_launch(void* const* d_in, const int* in_sizes, int n_in,
                              void* d_out, int out_size, void* d_ws, size_t ws_size,
                              hipStream_t stream)
{
  (void)in_sizes; (void)n_in; (void)out_size; (void)ws_size;
  // ---- inputs (jax pytree sorted-key flatten order) -----------------------
  // 0 z_t 1 node_v 2 edge_s 3 edge_v 4 edge_index 5 noise_level
  // 6..8   W_out:  wh, ws.b, ws.w
  // 9..14  We:     wh, ws.b, ws.w, wsv.b, wsv.w, wv
  // 15,16  We_ln:  b, w
  // 17..22 Wv:     wh, ws.b, ws.w, wsv.b, wsv.w, wv
  // 23,24  Wv_ln:  b, w
  // 25+34*i enc layer i: ff0(6) ff1(6) ln0(b,w) ln1(b,w) m0(6) m1(6) m2(6)
  // 161 mix.b 162 mix.w 163 mlp1.b 164 mlp1.w 165 mlp2.b 166 mlp2.w
  // 167 seq_res.b 168 seq_res.w 169 tfc1.b 170 tfc1.w 171 tfc2.b 172 tfc2.w
  // 173 time_w  174+16*i trans: fc1(b,w) fc2(b,w) k(b,w) ln1(b,w) ln2(b,w)
  //                              o(b,w) q(b,w) v(b,w)
  #define IN(i) ((const float*)d_in[(i)])
  const float* z_t   = IN(0);
  const float* nodev = IN(1);
  const float* edges = IN(2);
  const float* edgev = IN(3);
  const int*   eidx  = (const int*)d_in[4];
  const float* noise = IN(5);
  float* out = (float*)d_out;
  hipStream_t st = stream;

  // ---- workspace carve-up -------------------------------------------------
  char* wsb = (char*)d_ws; size_t off = 0;
  auto alloc = [&](size_t nfl) {
    float* p = (float*)(wsb + off);
    off += nfl * sizeof(float);
    off = (off + 255) & ~(size_t)255;
    return p;
  };
  float* sN    = alloc((size_t)Nn * 128);
  float* vN    = alloc((size_t)Nn * 48);
  float* X24   = alloc((size_t)Nn * 24);
  float* vT4   = alloc((size_t)Nn * 12);
  float* vT16  = alloc((size_t)Nn * 48);
  float* vhN   = alloc((size_t)Nn * 96);
  float* vhN2  = alloc((size_t)Nn * 96);
  float* vtmpN = alloc((size_t)Nn * 96);
  float* gateN = alloc((size_t)Nn * 32);
  float* XfN   = alloc((size_t)Nn * 160);
  float* Xf2N  = alloc((size_t)Nn * 544);
  float* fs2   = alloc((size_t)Nn * 128);
  float* fvN   = alloc((size_t)Nn * 48);
  float* XoN   = alloc((size_t)Nn * 144);
  float* Xmix  = alloc((size_t)Nn * 256);
  float* xb0   = alloc((size_t)Nn * 128);
  float* aggS  = alloc((size_t)Nn * 128);
  float* aggV  = alloc((size_t)Nn * 48);
  float* cnt   = alloc((size_t)Nn);
  float* esb   = alloc((size_t)Ee * 32);
  float* evb   = alloc((size_t)Ee * 3);
  float* XeE   = alloc((size_t)Ee * 33);
  // edge-chunk scratch
  float* Xcat  = alloc((size_t)CE * 321);
  float* X1c   = alloc((size_t)CE * 144);
  float* X2c   = alloc((size_t)CE * 144);
  float* smC   = alloc((size_t)CE * 128);
  float* vhC   = alloc((size_t)CE * 99);
  float* vtC   = alloc((size_t)CE * 99);
  float* gateC = alloc((size_t)CE * 33);
  float* mvTC  = alloc((size_t)CE * 99);
  float* mvF   = alloc((size_t)CE * 48);
  // transformer
  float* xT    = alloc((size_t)TOK * 128);
  float* Qb    = alloc((size_t)TOK * 128);
  float* Kb    = alloc((size_t)TOK * 128);
  float* Vb    = alloc((size_t)TOK * 128);
  float* Ob    = alloc((size_t)TOK * 128);
  float* tmpT  = alloc((size_t)TOK * 128);
  float* Fb    = alloc((size_t)TOK * 512);
  float* att   = alloc((size_t)128 * 512 * 512);
  float* tc    = alloc((size_t)B_ * 128);
  float* outT  = alloc((size_t)TOK * OUTD);

  // ====================== Phase A: node embedding (Wv) =====================
  k_node_init<<<gblk(Nn),256,0,st>>>(z_t, nodev, IN(24), IN(23), X24, vT4);
  gemm(st,1,0, vT4,4,0, IN(17),4,0, nullptr, vhN,16,0, 3*Nn,4,16, 1, 1.f);
  k_vnorm3<<<gblk((long)Nn*16),256,0,st>>>(vhN, X24, 24, 8, Nn, 16);
  gemm(st,1,0, X24,24,0, IN(19),24,0, IN(18), sN,128,0, Nn,24,128, 1, 1.f);
  gemm(st,1,0, vhN,16,0, IN(22),16,0, nullptr, vtmpN,16,0, 3*Nn,16,16, 1, 1.f);
  gemm(st,1,0, sN,128,0, IN(21),128,0, IN(20), gateN,16,0, Nn,128,16, 1, 1.f);
  k_gate_mul<<<gblk((long)Nn*48),256,0,st>>>(vtmpN, gateN, Nn, 16);
  k_transpose_out<<<gblk((long)Nn*48),256,0,st>>>(vtmpN, vN, Nn, 16);

  // ====================== Phase B: edge embedding (We) =====================
  k_edge_init<<<gblk(Ee),256,0,st>>>(edges, edgev, IN(16), IN(15), IN(9), XeE, evb);
  gemm(st,1,0, XeE,33,0, IN(11),33,0, IN(10), esb,32,0, Ee,33,32, 1, 1.f);
  k_edge_gate<<<gblk(Ee),256,0,st>>>(esb, IN(13), IN(12), IN(14), evb);

  // destination counts (same for all layers)
  k_zero<<<gblk(Nn),256,0,st>>>(cnt, Nn);
  k_count<<<gblk(Ee),256,0,st>>>(eidx + Ee, cnt);

  // ====================== Phase C: 4 GVP conv layers =======================
  for (int li = 0; li < 4; ++li) {
    const int base = 25 + 34 * li;
    k_zero<<<gblk((long)Nn*128),256,0,st>>>(aggS, (long)Nn*128);
    k_zero<<<gblk((long)Nn*48),256,0,st>>>(aggV, (long)Nn*48);
    for (int ch = 0; ch < NCH; ++ch) {
      const long e0 = (long)ch * CE;
      const int* src = eidx + e0;
      const int* dst = eidx + Ee + e0;
      k_gather_s<<<gblk((long)CE*288),256,0,st>>>(src, dst, sN, esb + e0*32, Xcat, CE);
      k_gather_v<<<gblk((long)CE*99),256,0,st>>>(src, dst, vN, evb + e0*3, mvTC, CE);
      // ---- m0 (si=288, vi=33, h=33, so=128, vo=16, relu) ----
      gemm(st,1,0, mvTC,33,0, IN(base+16),33,0, nullptr, vhC,33,0, 3*CE,33,33, 1, 1.f);
      k_vnorm3<<<gblk((long)CE*33),256,0,st>>>(vhC, Xcat, 321, 288, CE, 33);
      gemm(st,1,0, Xcat,321,0, IN(base+18),321,0, IN(base+17), X1c,144,0, CE,321,128, 1, 1.f);
      gemm(st,1,0, vhC,33,0, IN(base+21),33,0, nullptr, vtC,16,0, 3*CE,33,16, 1, 1.f);
      gemm(st,1,0, X1c,144,0, IN(base+20),128,0, IN(base+19), gateC,16,0, CE,128,16, 1, 1.f);
      k_gate_mul<<<gblk((long)CE*48),256,0,st>>>(vtC, gateC, CE, 16);
      k_relu_cols<<<gblk((long)CE*128),256,0,st>>>(X1c, 144, 128, CE);
      // ---- m1 (relu) ----
      gemm(st,1,0, vtC,16,0, IN(base+22),16,0, nullptr, vhC,16,0, 3*CE,16,16, 1, 1.f);
      k_vnorm3<<<gblk((long)CE*16),256,0,st>>>(vhC, X1c, 144, 128, CE, 16);
      gemm(st,1,0, X1c,144,0, IN(base+24),144,0, IN(base+23), X2c,144,0, CE,144,128, 1, 1.f);
      gemm(st,1,0, vhC,16,0, IN(base+27),16,0, nullptr, vtC,16,0, 3*CE,16,16, 1, 1.f);
      gemm(st,1,0, X2c,144,0, IN(base+26),128,0, IN(base+25), gateC,16,0, CE,128,16, 1, 1.f);
      k_gate_mul<<<gblk((long)CE*48),256,0,st>>>(vtC, gateC, CE, 16);
      k_relu_cols<<<gblk((long)CE*128),256,0,st>>>(X2c, 144, 128, CE);
      // ---- m2 (no act) ----
      gemm(st,1,0, vtC,16,0, IN(base+28),16,0, nullptr, vhC,16,0, 3*CE,16,16, 1, 1.f);
      k_vnorm3<<<gblk((long)CE*16),256,0,st>>>(vhC, X2c, 144, 128, CE, 16);
      gemm(st,1,0, X2c,144,0, IN(base+30),144,0, IN(base+29), smC,128,0, CE,144,128, 1, 1.f);
      gemm(st,1,0, vhC,16,0, IN(base+33),16,0, nullptr, vtC,16,0, 3*CE,16,16, 1, 1.f);
      gemm(st,1,0, smC,128,0, IN(base+32),128,0, IN(base+31), gateC,16,0, CE,128,16, 1, 1.f);
      k_gate_mul<<<gblk((long)CE*48),256,0,st>>>(vtC, gateC, CE, 16);
      k_transpose_out<<<gblk((long)CE*48),256,0,st>>>(vtC, mvF, CE, 16);
      // ---- mean aggregation ----
      k_scat_s<<<gblk((long)CE*128),256,0,st>>>(smC, dst, aggS, CE);
      k_scat_v<<<gblk((long)CE*48),256,0,st>>>(mvF, dst, aggV, CE);
    }
    // residual + gvp_ln (ln0)
    k_node_upd<<<gblk(Nn),256,0,st>>>(sN, vN, aggS, aggV, cnt, IN(base+13), IN(base+12));
    // ---- feed-forward GVPs: ff0 (relu) then ff1 ----
    k_transpose_in<<<gblk((long)Nn*48),256,0,st>>>(vN, vT16, Nn, 16);
    gemm(st,1,0, vT16,16,0, IN(base+0),16,0, nullptr, vhN,32,0, 3*Nn,16,32, 1, 1.f);
    k_copy_cols<<<gblk((long)Nn*128),256,0,st>>>(sN, 128, XfN, 160, 0, Nn);
    k_vnorm3<<<gblk((long)Nn*32),256,0,st>>>(vhN, XfN, 160, 128, Nn, 32);
    gemm(st,1,0, XfN,160,0, IN(base+2),160,0, IN(base+1), Xf2N,544,0, Nn,160,512, 1, 1.f);
    gemm(st,1,0, vhN,32,0, IN(base+5),32,0, nullptr, vtmpN,32,0, 3*Nn,32,32, 1, 1.f);
    gemm(st,1,0, Xf2N,544,0, IN(base+4),512,0, IN(base+3), gateN,32,0, Nn,512,32, 1, 1.f);
    k_gate_mul<<<gblk((long)Nn*96),256,0,st>>>(vtmpN, gateN, Nn, 32);
    k_relu_cols<<<gblk((long)Nn*512),256,0,st>>>(Xf2N, 544, 512, Nn);
    gemm(st,1,0, vtmpN,32,0, IN(base+6),32,0, nullptr, vhN2,32,0, 3*Nn,32,32, 1, 1.f);
    k_vnorm3<<<gblk((long)Nn*32),256,0,st>>>(vhN2, Xf2N, 544, 512, Nn, 32);
    gemm(st,1,0, Xf2N,544,0, IN(base+8),544,0, IN(base+7), fs2,128,0, Nn,544,128, 1, 1.f);
    gemm(st,1,0, vhN2,32,0, IN(base+11),32,0, nullptr, vtmpN,16,0, 3*Nn,32,16, 1, 1.f);
    gemm(st,1,0, fs2,128,0, IN(base+10),128,0, IN(base+9), gateN,16,0, Nn,128,16, 1, 1.f);
    k_gate_mul<<<gblk((long)Nn*48),256,0,st>>>(vtmpN, gateN, Nn, 16);
    k_transpose_out<<<gblk((long)Nn*48),256,0,st>>>(vtmpN, fvN, Nn, 16);
    // residual + gvp_ln (ln1)
    k_node_upd<<<gblk(Nn),256,0,st>>>(sN, vN, fs2, fvN, nullptr, IN(base+15), IN(base+14));
  }

  // ====================== Phase D: readout + mixing ========================
  k_transpose_in<<<gblk((long)Nn*48),256,0,st>>>(vN, vT16, Nn, 16);
  gemm(st,1,0, vT16,16,0, IN(6),16,0, nullptr, vhN,16,0, 3*Nn,16,16, 1, 1.f);
  k_copy_cols<<<gblk((long)Nn*128),256,0,st>>>(sN, 128, XoN, 144, 0, Nn);
  k_vnorm3<<<gblk((long)Nn*16),256,0,st>>>(vhN, XoN, 144, 128, Nn, 16);
  gemm(st,1,0, XoN,144,0, IN(8),144,0, IN(7), Xmix,256,0, Nn,144,128, 1, 1.f);
  // seq_res on raw init_seq = [z_t, 0]: only first 4 weight cols matter
  gemm(st,1,0, z_t,4,0, IN(168),8,0, IN(167), Xmix + 128,256,0, Nn,4,128, 1, 1.f);
  gemm(st,1,0, Xmix,256,0, IN(162),256,0, IN(161), xb0,128,0, Nn,256,128, 1, 1.f);
  k_time<<<dim3(B_),dim3(256),0,st>>>(noise, IN(173), IN(169), IN(170), IN(171), IN(172), tc);
  k_build_x<<<gblk((long)TOK*128),256,0,st>>>(xb0, xT);

  // ====================== Phase E: 4 transformer layers ====================
  for (int ti = 0; ti < 4; ++ti) {
    const int tb = 174 + 16 * ti;
    k_add_cond<<<gblk((long)TOK*128),256,0,st>>>(xT, tc);
    gemm(st,1,0, xT,128,0, IN(tb+13),128,0, IN(tb+12), Qb,128,0, TOK,128,128, 1, 0.25f); // q*1/sqrt(hd)
    gemm(st,1,0, xT,128,0, IN(tb+5),128,0, IN(tb+4),  Kb,128,0, TOK,128,128, 1, 1.f);
    gemm(st,1,0, xT,128,0, IN(tb+15),128,0, IN(tb+14),Vb,128,0, TOK,128,128, 1, 1.f);
    // scores: batched over z=b*8+h (base offset 16*z, strides 16)
    gemm(st,1,0, Qb,2048,16, Kb,2048,16, nullptr, att,512,(long)512*512,
         512,HD,512, 128, 1.f);
    k_softmax512<<<gblk((long)128*512),256,0,st>>>(att, (long)128*512);
    // O = att @ V  (B as [K,N])
    gemm(st,0,0, att,512,(long)512*512, Vb,2048,16, nullptr, Ob,2048,16,
         512,512,HD, 128, 1.f);
    gemm(st,1,0, Ob,128,0, IN(tb+11),128,0, IN(tb+10), tmpT,128,0, TOK,128,128, 1, 1.f);
    k_add<<<gblk((long)TOK*128),256,0,st>>>(xT, tmpT, (long)TOK*128);
    k_layernorm<<<gblk(TOK),256,0,st>>>(xT, IN(tb+7), IN(tb+6), TOK, 128);
    gemm(st,1,1, xT,128,0, IN(tb+1),128,0, IN(tb+0), Fb,512,0, TOK,128,512, 1, 1.f); // relu
    gemm(st,1,0, Fb,512,0, IN(tb+3),512,0, IN(tb+2), tmpT,128,0, TOK,512,128, 1, 1.f);
    k_add<<<gblk((long)TOK*128),256,0,st>>>(xT, tmpT, (long)TOK*128);
    k_layernorm<<<gblk(TOK),256,0,st>>>(xT, IN(tb+9), IN(tb+8), TOK, 128);
  }

  // ====================== Phase F: output MLP ==============================
  gemm(st,1,1, xT,128,0, IN(164),128,0, IN(163), Qb,128,0, TOK,128,128, 1, 1.f); // relu
  gemm(st,1,0, Qb,128,0, IN(166),128,0, IN(165), outT,OUTD,0, TOK,128,OUTD, 1, 1.f);
  k_final<<<gblk((long)TOK*OUTD),256,0,st>>>(outT, out);
}
